// Wav2Vec2BertSelfAttention_13451837571617
// MI455X (gfx1250) — compile-verified, matrix operand order checked by reading, not run on hardware
//
#include <hip/hip_runtime.h>
#include <hip/hip_bf16.h>
#include <math.h>

typedef __bf16 bf16_t;
typedef __attribute__((ext_vector_type(16))) bf16_t v16bf;
typedef __attribute__((ext_vector_type(8)))  bf16_t v8bf;
typedef __attribute__((ext_vector_type(8)))  float  v8f;
typedef __attribute__((ext_vector_type(4)))  float  v4f;

#define WMMA_BF16(A, B, C) \
  __builtin_amdgcn_wmma_f32_16x16x32_bf16(false, (A), false, (B), (short)0, (C), false, false)

// ---------------------------------------------------------------------------
// Fragment loaders.
// A/B 16-bit fragment layout (ISA 7.12.2): lane i = lane&15 owns row/col i.
// hf = lane>>4 selects K-groups: e0..7 -> K = 8*hf + e ; e8..15 -> K = 16 + 8*hf + (e-8).
// `p` points at this lane's row start at k0 (row-major, contiguous in K).
// ---------------------------------------------------------------------------
static __device__ inline v16bf frag_from_f32(const float* __restrict__ p, int hf) {
  const v4f* q0 = (const v4f*)(p + hf * 8);
  const v4f* q1 = (const v4f*)(p + 16 + hf * 8);
  v4f x0 = q0[0], x1 = q0[1], x2 = q1[0], x3 = q1[1];
  v16bf f;
#pragma unroll
  for (int e = 0; e < 4; ++e) {
    f[e]      = (bf16_t)x0[e];
    f[e + 4]  = (bf16_t)x1[e];
    f[e + 8]  = (bf16_t)x2[e];
    f[e + 12] = (bf16_t)x3[e];
  }
  return f;
}

static __device__ inline v16bf frag_from_bf16(const bf16_t* __restrict__ p, int hf) {
  v8bf a = *(const v8bf*)(p + hf * 8);
  v8bf b = *(const v8bf*)(p + 16 + hf * 8);
  v16bf f;
#pragma unroll
  for (int e = 0; e < 8; ++e) { f[e] = a[e]; f[e + 8] = b[e]; }
  return f;
}

// ---------------------------------------------------------------------------
// K0a: row-wise fp32 -> bf16 (for hidden_states and rel-pos embeddings)
// ---------------------------------------------------------------------------
__global__ __launch_bounds__(256) void f32_to_bf16_kernel(const float* __restrict__ src,
                                                          bf16_t* __restrict__ dst, int n8) {
  int idx = blockIdx.x * 256 + threadIdx.x;
  if (idx >= n8) return;
  const v4f* s = (const v4f*)(src + (size_t)idx * 8);
  v4f a = s[0], b = s[1];
  v8bf o;
#pragma unroll
  for (int e = 0; e < 4; ++e) { o[e] = (bf16_t)a[e]; o[e + 4] = (bf16_t)b[e]; }
  *(v8bf*)(dst + (size_t)idx * 8) = o;
}

// ---------------------------------------------------------------------------
// K0b: fp32 [1024 x 1024] W  ->  bf16 W^T [1024 x 1024]  (32x32 LDS transpose)
// ---------------------------------------------------------------------------
__global__ __launch_bounds__(256) void wcvt_kernel(const float* __restrict__ W,
                                                   bf16_t* __restrict__ WT) {
  __shared__ float tile[32][33];
  int tx = threadIdx.x & 31, ty = threadIdx.x >> 5;
  int bx = blockIdx.x, by = blockIdx.y;
#pragma unroll
  for (int i = 0; i < 4; ++i)
    tile[ty + i * 8][tx] = W[(by * 32 + ty + i * 8) * 1024 + bx * 32 + tx];
  __syncthreads();
#pragma unroll
  for (int i = 0; i < 4; ++i)
    WT[(bx * 32 + ty + i * 8) * 1024 + by * 32 + tx] = (bf16_t)tile[tx][ty + i * 8];
}

// ---------------------------------------------------------------------------
// K1: GEMM  C[M x 1024] = A[M x 1024] (bf16) * WT^T (bf16, WT is [N][K]) + bias
// One wave = 16 x 64 strip (4 WMMA tiles, A fragment reused 4x); 8 waves/block.
// Ping-pong double buffering (x2 unrolled K-loop): fragment sets A/B alternate
// as load destination vs WMMA source, so no register-rotation copies.
// mode 0: fp32 natural C[m*1024 + n]
// mode 2: bf16 natural
// mode 3: bf16 scatter to V^T  ([B, H, HD, L] so probs@V reads contiguous K)
// ---------------------------------------------------------------------------
__global__ __launch_bounds__(256) void gemm_bf16_wmma(const bf16_t* __restrict__ A,
                                                      const bf16_t* __restrict__ WT,
                                                      const float* __restrict__ bias,
                                                      void* __restrict__ Cout,
                                                      int M, int mode) {
  const int K = 1024;
  int lane = threadIdx.x & 31;
  int wave = threadIdx.x >> 5;
  int wid  = blockIdx.x * 8 + wave;
  int tilesM = (M + 15) >> 4;
  if (wid >= tilesM * 16) return;              // wave-uniform: EXEC stays all-1s
  int tm = wid >> 4, tn4 = wid & 15;
  int i = lane & 15, hf = lane >> 4;

  int row  = tm * 16 + i;
  int rowc = row < M ? row : (M - 1);          // clamp loads; stores are guarded
  const bf16_t* aP = A + (size_t)rowc * K;
  const bf16_t* bP[4];
#pragma unroll
  for (int j = 0; j < 4; ++j) bP[j] = WT + (size_t)(tn4 * 64 + j * 16 + i) * K;

  v8f acc[4] = {};
  // prologue: set A @ k0=0
  v16bf afA = frag_from_bf16(aP, hf);
  v16bf bfA[4];
#pragma unroll
  for (int j = 0; j < 4; ++j) bfA[j] = frag_from_bf16(bP[j], hf);

  for (int k0 = 0; k0 < K - 64; k0 += 64) {
    // load set B @ k0+32, compute with set A
    v16bf afB = frag_from_bf16(aP + k0 + 32, hf);
    v16bf bfB[4];
#pragma unroll
    for (int j = 0; j < 4; ++j) bfB[j] = frag_from_bf16(bP[j] + k0 + 32, hf);
#pragma unroll
    for (int j = 0; j < 4; ++j) acc[j] = WMMA_BF16(afA, bfA[j], acc[j]);
    // load set A @ k0+64, compute with set B
    afA = frag_from_bf16(aP + k0 + 64, hf);
#pragma unroll
    for (int j = 0; j < 4; ++j) bfA[j] = frag_from_bf16(bP[j] + k0 + 64, hf);
#pragma unroll
    for (int j = 0; j < 4; ++j) acc[j] = WMMA_BF16(afB, bfB[j], acc[j]);
  }
  {  // tail: k0 = K-64 (set A loaded), then K-32 (set B)
    v16bf afB = frag_from_bf16(aP + K - 32, hf);
    v16bf bfB[4];
#pragma unroll
    for (int j = 0; j < 4; ++j) bfB[j] = frag_from_bf16(bP[j] + K - 32, hf);
#pragma unroll
    for (int j = 0; j < 4; ++j) acc[j] = WMMA_BF16(afA, bfA[j], acc[j]);
#pragma unroll
    for (int j = 0; j < 4; ++j) acc[j] = WMMA_BF16(afB, bfB[j], acc[j]);
  }

#pragma unroll
  for (int j = 0; j < 4; ++j) {
    int n = tn4 * 64 + j * 16 + i;
    float bv = bias ? bias[n] : 0.0f;
#pragma unroll
    for (int r = 0; r < 8; ++r) {
      int m = tm * 16 + r + 8 * hf;
      if (m < M) {
        float val = acc[j][r] + bv;
        if (mode == 0) {
          ((float*)Cout)[(size_t)m * 1024 + n] = val;
        } else if (mode == 2) {
          ((bf16_t*)Cout)[(size_t)m * 1024 + n] = (bf16_t)val;
        } else {                               // mode 3: V^T bf16
          int b = m >> 10, l = m & 1023;
          int h = n >> 6,  d = n & 63;
          ((bf16_t*)Cout)[(size_t)(((b * 16 + h) * 64 + d) << 10) + l] = (bf16_t)val;
        }
      }
    }
  }
}

// ---------------------------------------------------------------------------
// K2: fused scores (content + shifted positional) + softmax -> probs.
// One block per (b, h, 16-row l-block). 8 waves.
//   sbd[l, m] = qv[l] . pos[1023 + m - l]
//   band G[16][1040]: G[l - l0][j] = qv[l] . pos[(1008 - l0) + j]
//   => sbd[l, m] = G[l - l0][m + 15 - (l - l0)]
// ---------------------------------------------------------------------------
__global__ __launch_bounds__(256) void scores_softmax_kernel(
    const float*  __restrict__ q,      // fp32 [B, L, H, HD]
    const bf16_t* __restrict__ kmat,   // bf16 [B, L, H, HD]
    const bf16_t* __restrict__ pos,    // bf16 [2048, 1024], row 2047 = zeros
    const float*  __restrict__ ubias,  // [H, HD]
    const float*  __restrict__ vbias,  // [H, HD]
    float* __restrict__ probs) {       // fp32 [B, H, L, L]
  __shared__ bf16_t Gs[16][1056];      // 1040 used (+pad)
  __shared__ float red[2][8][16];

  int bid = blockIdx.x;
  int lt = bid & 63;
  int h  = (bid >> 6) & 15;
  int bI = bid >> 10;
  int l0 = lt * 16;

  int lane = threadIdx.x & 31;
  int wave = threadIdx.x >> 5;
  int i = lane & 15, hf = lane >> 4;

  // --- build q+u and q+v A-fragments for row l0+i (kept in registers) ---
  const float* qrow = q + (size_t)((bI * 1024 + l0 + i) * 1024) + h * 64;
  const float* urow = ubias + h * 64;
  const float* vrow = vbias + h * 64;
  v16bf qu[2], qv[2];
#pragma unroll
  for (int s = 0; s < 2; ++s) {
    int ks = s * 32;
    v16bf fu, fv;
#pragma unroll
    for (int e = 0; e < 16; ++e) {
      int de = ks + ((e >> 3) << 4) + hf * 8 + (e & 7);
      float qq = qrow[de];
      fu[e] = (bf16_t)(qq + urow[de]);
      fv[e] = (bf16_t)(qq + vrow[de]);
    }
    qu[s] = fu; qv[s] = fv;
  }

  // --- positional band G via WMMA, 65 column tiles split across 8 waves ---
  int p_lo = 1008 - l0;                        // >= 0; p_lo + 1039 <= 2047 (pad row)
  for (int jt = wave; jt < 65; jt += 8) {      // wave-uniform trip counts
    const bf16_t* prow = pos + (size_t)(p_lo + jt * 16 + i) * 1024 + h * 64;
    v8f acc = {};
#pragma unroll
    for (int s = 0; s < 2; ++s) {
      v16bf bfrag = frag_from_bf16(prow + s * 32, hf);
      acc = WMMA_BF16(qv[s], bfrag, acc);
    }
#pragma unroll
    for (int r = 0; r < 8; ++r) Gs[r + 8 * hf][jt * 16 + i] = (bf16_t)acc[r];
  }
  __syncthreads();

  // --- content scores + shifted band add; wave owns cols [wave*128, +128) ---
  const float scale = 0.125f;                  // 1/sqrt(64)
  const bf16_t* kbase = kmat + (size_t)(bI * 1024) * 1024 + h * 64;
  v8f sacc[8];
  // prologue: K fragments for tile t=0 (loop below is fully unrolled; SSA
  // renaming removes the rotation copies)
  v16bf kf0 = frag_from_bf16(kbase + (size_t)(wave * 128 + i) * 1024, hf);
  v16bf kf1 = frag_from_bf16(kbase + (size_t)(wave * 128 + i) * 1024 + 32, hf);
#pragma unroll
  for (int t = 0; t < 8; ++t) {
    v16bf nf0 = kf0, nf1 = kf1;
    if (t < 7) {                               // prefetch next m-tile's K frags
      const bf16_t* krow = kbase + (size_t)(wave * 128 + (t + 1) * 16 + i) * 1024;
      nf0 = frag_from_bf16(krow, hf);
      nf1 = frag_from_bf16(krow + 32, hf);
    }
    v8f acc = {};
    acc = WMMA_BF16(qu[0], kf0, acc);
    acc = WMMA_BF16(qu[1], kf1, acc);
    int m = wave * 128 + t * 16 + i;
#pragma unroll
    for (int r = 0; r < 8; ++r) {
      int lr = r + 8 * hf;
      float bd = (float)Gs[lr][m + 15 - lr];
      acc[r] = (acc[r] + bd) * scale;
    }
    sacc[t] = acc;
    kf0 = nf0; kf1 = nf1;
  }

  // --- softmax: per-row max (shuffle within 16-lane half, then LDS x-wave) ---
  float rmax[8];
#pragma unroll
  for (int r = 0; r < 8; ++r) {
    float mx = -1e30f;
#pragma unroll
    for (int t = 0; t < 8; ++t) mx = fmaxf(mx, sacc[t][r]);
#pragma unroll
    for (int o = 8; o >= 1; o >>= 1) mx = fmaxf(mx, __shfl_xor(mx, o, 32));
    rmax[r] = mx;
  }
  if (i == 0) {
#pragma unroll
    for (int r = 0; r < 8; ++r) red[0][wave][r + 8 * hf] = rmax[r];
  }
  __syncthreads();
  float gmax[8];
#pragma unroll
  for (int r = 0; r < 8; ++r) {
    int lr = r + 8 * hf;
    float mx = -1e30f;
#pragma unroll
    for (int w = 0; w < 8; ++w) mx = fmaxf(mx, red[0][w][lr]);
    gmax[r] = mx;
  }

  // --- exp + per-row sum ---
  float rsum[8];
#pragma unroll
  for (int r = 0; r < 8; ++r) {
    float s = 0.0f;
#pragma unroll
    for (int t = 0; t < 8; ++t) {
      float e = __expf(sacc[t][r] - gmax[r]);
      sacc[t][r] = e;
      s += e;
    }
#pragma unroll
    for (int o = 8; o >= 1; o >>= 1) s += __shfl_xor(s, o, 32);
    rsum[r] = s;
  }
  if (i == 0) {
#pragma unroll
    for (int r = 0; r < 8; ++r) red[1][wave][r + 8 * hf] = rsum[r];
  }
  __syncthreads();
  float ginv[8];
#pragma unroll
  for (int r = 0; r < 8; ++r) {
    int lr = r + 8 * hf;
    float s = 0.0f;
#pragma unroll
    for (int w = 0; w < 8; ++w) s += red[1][w][lr];
    ginv[r] = 1.0f / s;
  }

  // --- write probs (fp32, second output) ---
  float* pbase = probs + (size_t)((bI * 16 + h) * 1024) * 1024;
#pragma unroll
  for (int t = 0; t < 8; ++t) {
    int m = wave * 128 + t * 16 + i;
#pragma unroll
    for (int r = 0; r < 8; ++r) {
      int lr = r + 8 * hf;
      pbase[(size_t)(l0 + lr) * 1024 + m] = sacc[t][r] * ginv[r];
    }
  }
}

// ---------------------------------------------------------------------------
// K3: ctx[b, l, h, d] = sum_m probs[b,h,l,m] * v[b,m,h,d]   (V^T bf16)
// One wave = 16(l) x 64(d = full head) strip, K = 1024 over m; A reused 4x.
// Ping-pong double buffered like K1. ctx stored bf16 (feeds final WMMA GEMM).
// ---------------------------------------------------------------------------
__global__ __launch_bounds__(256) void ctx_gemm_kernel(const float*  __restrict__ probs,
                                                       const bf16_t* __restrict__ vT,  // [B,H,HD,L]
                                                       bf16_t* __restrict__ ctx) {     // [B,L,H,HD]
  int lane = threadIdx.x & 31;
  int wid  = blockIdx.x * 8 + (threadIdx.x >> 5);
  int i = lane & 15, hf = lane >> 4;
  int lt = wid & 63;
  int h  = (wid >> 6) & 15;
  int bI = wid >> 10;

  const float* arow = probs + (size_t)((bI * 16 + h) * 1024 + lt * 16 + i) * 1024;
  const bf16_t* brow[4];
#pragma unroll
  for (int j = 0; j < 4; ++j)
    brow[j] = vT + (size_t)((bI * 16 + h) * 64 + j * 16 + i) * 1024;

  v8f acc[4] = {};
  v16bf afA = frag_from_f32(arow, hf);         // probs live in fp32 output buffer
  v16bf bfA[4];
#pragma unroll
  for (int j = 0; j < 4; ++j) bfA[j] = frag_from_bf16(brow[j], hf);

  for (int k0 = 0; k0 < 1024 - 64; k0 += 64) {
    v16bf afB = frag_from_f32(arow + k0 + 32, hf);
    v16bf bfB[4];
#pragma unroll
    for (int j = 0; j < 4; ++j) bfB[j] = frag_from_bf16(brow[j] + k0 + 32, hf);
#pragma unroll
    for (int j = 0; j < 4; ++j) acc[j] = WMMA_BF16(afA, bfA[j], acc[j]);
    afA = frag_from_f32(arow + k0 + 64, hf);
#pragma unroll
    for (int j = 0; j < 4; ++j) bfA[j] = frag_from_bf16(brow[j] + k0 + 64, hf);
#pragma unroll
    for (int j = 0; j < 4; ++j) acc[j] = WMMA_BF16(afB, bfB[j], acc[j]);
  }
  {
    v16bf afB = frag_from_f32(arow + 1024 - 32, hf);
    v16bf bfB[4];
#pragma unroll
    for (int j = 0; j < 4; ++j) bfB[j] = frag_from_bf16(brow[j] + 1024 - 32, hf);
#pragma unroll
    for (int j = 0; j < 4; ++j) acc[j] = WMMA_BF16(afA, bfA[j], acc[j]);
#pragma unroll
    for (int j = 0; j < 4; ++j) acc[j] = WMMA_BF16(afB, bfB[j], acc[j]);
  }

#pragma unroll
  for (int j = 0; j < 4; ++j) {
#pragma unroll
    for (int r = 0; r < 8; ++r) {
      int l = lt * 16 + r + 8 * hf;
      ctx[(size_t)(bI * 1024 + l) * 1024 + h * 64 + j * 16 + i] = (bf16_t)acc[j][r];
    }
  }
}

// ---------------------------------------------------------------------------
// Host launch
// ---------------------------------------------------------------------------
extern "C" void kernel_launch(void* const* d_in, const int* in_sizes, int n_in,
                              void* d_out, int out_size, void* d_ws, size_t ws_size,
                              hipStream_t stream) {
  (void)in_sizes; (void)n_in; (void)out_size; (void)ws_size;

  const float* X      = (const float*)d_in[0];   // [4,1024,1024]
  const float* relpos = (const float*)d_in[1];   // [1,2047,1024]
  const float* Wq = (const float*)d_in[2];  const float* bq = (const float*)d_in[3];
  const float* Wk = (const float*)d_in[4];  const float* bk = (const float*)d_in[5];
  const float* Wv = (const float*)d_in[6];  const float* bv = (const float*)d_in[7];
  const float* Wpos = (const float*)d_in[8];
  const float* pu = (const float*)d_in[9];  const float* pv = (const float*)d_in[10];
  const float* Wo = (const float*)d_in[11]; const float* bo = (const float*)d_in[12];

  float* outp  = (float*)d_out;                            // [4,1024,1024]
  float* probs = (float*)d_out + (size_t)4 * 1024 * 1024;  // [4,16,1024,1024]

  char* ws = (char*)d_ws;
  size_t off = 0;
  float*  qb    = (float*)(ws + off);  off += (size_t)4 * 1024 * 1024 * 4;  // fp32 q
  bf16_t* kb    = (bf16_t*)(ws + off); off += (size_t)4 * 1024 * 1024 * 2;  // bf16 k
  bf16_t* vTb   = (bf16_t*)(ws + off); off += (size_t)4 * 1024 * 1024 * 2;  // bf16 V^T
  bf16_t* posb  = (bf16_t*)(ws + off); off += (size_t)2048 * 1024 * 2;      // bf16, padded
  bf16_t* ctxb  = (bf16_t*)(ws + off); off += (size_t)4 * 1024 * 1024 * 2;  // bf16 ctx
  bf16_t* Xbf   = (bf16_t*)(ws + off); off += (size_t)4 * 1024 * 1024 * 2;  // bf16 X
  bf16_t* relbf = (bf16_t*)(ws + off); off += (size_t)2048 * 1024 * 2;      // bf16 relpos
  bf16_t* WqT   = (bf16_t*)(ws + off); off += (size_t)1024 * 1024 * 2;
  bf16_t* WkT   = (bf16_t*)(ws + off); off += (size_t)1024 * 1024 * 2;
  bf16_t* WvT   = (bf16_t*)(ws + off); off += (size_t)1024 * 1024 * 2;
  bf16_t* WposT = (bf16_t*)(ws + off); off += (size_t)1024 * 1024 * 2;
  bf16_t* WoT   = (bf16_t*)(ws + off); off += (size_t)1024 * 1024 * 2;

  dim3 blk(256);

  // K0: one-time operand conversions (amortized across all WMMA consumers)
  f32_to_bf16_kernel<<<2048, blk, 0, stream>>>(X, Xbf, 4 * 1024 * 1024 / 8);
  f32_to_bf16_kernel<<<1024, blk, 0, stream>>>(relpos, relbf, 2047 * 1024 / 8);
  wcvt_kernel<<<dim3(32, 32), blk, 0, stream>>>(Wq,   WqT);
  wcvt_kernel<<<dim3(32, 32), blk, 0, stream>>>(Wk,   WkT);
  wcvt_kernel<<<dim3(32, 32), blk, 0, stream>>>(Wv,   WvT);
  wcvt_kernel<<<dim3(32, 32), blk, 0, stream>>>(Wpos, WposT);
  wcvt_kernel<<<dim3(32, 32), blk, 0, stream>>>(Wo,   WoT);

  // zero the pos pad row (index 2047) used by the banded bd GEMM
  hipMemsetAsync(posb + (size_t)2047 * 1024, 0, 1024 * sizeof(bf16_t), stream);

  // K1: projections. q fp32 (bias-added + converted once in K2);
  //     k, pos bf16 natural; v bf16 transposed to [B,H,HD,L].
  gemm_bf16_wmma<<<512, blk, 0, stream>>>(Xbf, WqT, bq, qb, 4096, 0);
  gemm_bf16_wmma<<<512, blk, 0, stream>>>(Xbf, WkT, bk, kb, 4096, 2);
  gemm_bf16_wmma<<<512, blk, 0, stream>>>(Xbf, WvT, bv, vTb, 4096, 3);
  gemm_bf16_wmma<<<256, blk, 0, stream>>>(relbf, WposT, nullptr, posb, 2047, 2);

  // K2: fused scores + rel-shift + softmax -> probs (second output)
  scores_softmax_kernel<<<4096, blk, 0, stream>>>(qb, kb, posb, pu, pv, probs);

  // K3: ctx = probs @ v   (bf16 out, feeds final GEMM)
  ctx_gemm_kernel<<<512, blk, 0, stream>>>(probs, vTb, ctxb);

  // K4: out = ctx @ Wo + bo (first output)
  gemm_bf16_wmma<<<512, blk, 0, stream>>>(ctxb, WoT, bo, outp, 4096, 0);
}
